// CWNET_55319178772973
// MI455X (gfx1250) — compile-verified, matrix operand order checked by reading, not run on hardware
//
#include <hip/hip_runtime.h>
#include <hip/hip_bf16.h>
#include <math.h>

typedef __attribute__((ext_vector_type(16))) _Float16 v16h;
typedef __attribute__((ext_vector_type(8)))  float    v8f;

static constexpr int Bn = 8;
static constexpr int Np = 2048;
static constexpr int Kn = 20;
#define EPSBN 1e-5f

// ---------------------------------------------------------------- kNN top-k
__global__ __launch_bounds__(256)
void knn_kernel(const float* __restrict__ x, int C, int k, int* __restrict__ idx) {
  __shared__ float dist[2048];
  __shared__ float ctr[128];
  __shared__ float rv[256];
  __shared__ int   ri[256];
  int bn = blockIdx.x; int b = bn >> 11; int n = bn & 2047;
  const float* xb = x + (long)b * C * Np;
  for (int c = threadIdx.x; c < C; c += 256) ctr[c] = xb[(long)c * Np + n];
  __syncthreads();
  for (int m = threadIdx.x; m < Np; m += 256) {
    float d = 0.f;
    for (int c = 0; c < C; ++c) { float t = xb[(long)c * Np + m] - ctr[c]; d += t * t; }
    dist[m] = d;
  }
  __syncthreads();
  for (int it = 0; it < k; ++it) {
    float best = 3.4e38f; int bi = 0x7fffffff;
    for (int m = threadIdx.x; m < Np; m += 256) {
      float d = dist[m];
      if (d < best || (d == best && m < bi)) { best = d; bi = m; }
    }
    rv[threadIdx.x] = best; ri[threadIdx.x] = bi;
    __syncthreads();
    for (int s = 128; s > 0; s >>= 1) {
      if (threadIdx.x < s) {
        float ov = rv[threadIdx.x + s]; int oi = ri[threadIdx.x + s];
        if (ov < rv[threadIdx.x] || (ov == rv[threadIdx.x] && oi < ri[threadIdx.x])) {
          rv[threadIdx.x] = ov; ri[threadIdx.x] = oi;
        }
      }
      __syncthreads();
    }
    if (threadIdx.x == 0) { idx[(long)bn * k + it] = ri[0]; dist[ri[0]] = 3.4e38f; }
    __syncthreads();
  }
}

// ---------------------------------------------------------------- geometric descriptor
__global__ void geom_desc_kernel(const float* __restrict__ x, const int* __restrict__ idx3,
                                 float* __restrict__ xd) {
  int t = blockIdx.x * 256 + threadIdx.x; if (t >= Bn * Np) return;
  int b = t / Np, n = t % Np;
  const float* xb = x + (long)b * 3 * Np;
  int i1 = idx3[(long)t * 3 + 1], i2 = idx3[(long)t * 3 + 2];
  float cx = xb[n], cy = xb[Np + n], cz = xb[2 * Np + n];
  float e1x = xb[i1] - cx, e1y = xb[Np + i1] - cy, e1z = xb[2 * Np + i1] - cz;
  float e2x = xb[i2] - cx, e2y = xb[Np + i2] - cy, e2z = xb[2 * Np + i2] - cz;
  float nx = e1y * e2z - e1z * e2y;
  float ny = e1z * e2x - e1x * e2z;
  float nz = e1x * e2y - e1y * e2x;
  float d1 = sqrtf(e1x * e1x + e1y * e1y + e1z * e1z);
  float d2 = sqrtf(e2x * e2x + e2y * e2y + e2z * e2z);
  float* ob = xd + (long)b * 8 * Np;
  ob[0 * Np + n] = cx; ob[1 * Np + n] = cy; ob[2 * Np + n] = cz;
  ob[3 * Np + n] = nx; ob[4 * Np + n] = ny; ob[5 * Np + n] = nz;
  ob[6 * Np + n] = d1; ob[7 * Np + n] = d2;
}

// ---------------------------------------------------------------- graph feature / gather
__global__ void graph_feature_kernel(const float* __restrict__ feat, const int* __restrict__ idx,
                                     float* __restrict__ out, int C) {
  long t = (long)blockIdx.x * 256 + threadIdx.x;
  long tot = (long)Bn * C * Np * Kn; if (t >= tot) return;
  int j = (int)(t % Kn); long r = t / Kn;
  int n = (int)(r % Np); long r2 = r / Np;
  int c = (int)(r2 % C); int b = (int)(r2 / C);
  int m = idx[((long)b * Np + n) * Kn + j];
  float xc = feat[((long)b * C + c) * Np + n];
  float nb = feat[((long)b * C + c) * Np + m];
  long obase = (((long)b * 2 * C + c) * Np + n) * Kn + j;
  out[obase] = nb - xc;
  out[obase + (long)C * Np * Kn] = xc;
}

__global__ void gather_kernel(const float* __restrict__ feat, const int* __restrict__ idx,
                              float* __restrict__ out, int C, int mode) {
  long t = (long)blockIdx.x * 256 + threadIdx.x;
  long tot = (long)Bn * C * Np * Kn; if (t >= tot) return;
  int j = (int)(t % Kn); long r = t / Kn;
  int n = (int)(r % Np); long r2 = r / Np;
  int c = (int)(r2 % C); int b = (int)(r2 / C);
  int m = idx[((long)b * Np + n) * Kn + j];
  float nb = feat[((long)b * C + c) * Np + m];
  if (mode) { float ct = feat[((long)b * C + c) * Np + n]; out[t] = ct - nb; }
  else out[t] = nb;
}

// ---------------------------------------------------------------- batchnorm (training mode)
__global__ __launch_bounds__(256)
void bn_stats_kernel(const float* __restrict__ x, int C, long P,
                     float* __restrict__ mean, float* __restrict__ istd) {
  __shared__ float ss[256], s2[256];
  int c = blockIdx.x;
  long tot = (long)Bn * P;
  float a = 0.f, q = 0.f;
  for (long i = threadIdx.x; i < tot; i += 256) {
    long b = i / P, p = i - b * P;
    float v = x[(b * C + c) * P + p];
    a += v; q += v * v;
  }
  ss[threadIdx.x] = a; s2[threadIdx.x] = q;
  __syncthreads();
  for (int s = 128; s > 0; s >>= 1) {
    if (threadIdx.x < s) { ss[threadIdx.x] += ss[threadIdx.x + s]; s2[threadIdx.x] += s2[threadIdx.x + s]; }
    __syncthreads();
  }
  if (threadIdx.x == 0) {
    float m = ss[0] / (float)tot;
    float var = s2[0] / (float)tot - m * m;
    mean[c] = m; istd[c] = rsqrtf(fmaxf(var, 0.f) + EPSBN);
  }
}

__global__ void bn_apply_kernel(float* __restrict__ x, int C, long P,
                                const float* __restrict__ mean, const float* __restrict__ istd,
                                const float* __restrict__ g, const float* __restrict__ bt, int act) {
  long t = (long)blockIdx.x * 256 + threadIdx.x;
  long tot = (long)Bn * C * P; if (t >= tot) return;
  int c = (int)((t / P) % C);
  float v = (x[t] - mean[c]) * istd[c] * g[c] + bt[c];
  if (act == 1) v = v > 0.f ? v : 0.2f * v;
  else if (act == 2) v = fmaxf(v, 0.f);
  x[t] = v;
}

// ---------------------------------------------------------------- WMMA GEMM  C = alpha*op(A)op(B)+bias
// Block tile 64(M) x 128(N) x 32(K). 8 wave32 waves: wave w -> M-rows [ (w>>1)*16 ),
// N-columns [(w&1)*64 .. +64) as four 16-wide sub-tiles (4 accumulators, A fragment reused).
// LDS holds f16 K-pairs packed in 32-bit words so every fragment register is ONE ds_load_b32.
union PairPK { _Float16 h[2]; unsigned u; };
union FragU  { v16h v; unsigned u[8]; };

__global__ __launch_bounds__(256)
void gemm_wmma_kernel(const float* __restrict__ A, const float* __restrict__ Bm,
                      float* __restrict__ C, const float* __restrict__ bias,
                      int M, int N, int K, long sA, long sB, long sC,
                      int lda, int ldb, int ldc, int tA, int tB, float alpha) {
  __shared__ unsigned Ap[64][18];   // [m][kpair]   (pad 16->18)
  __shared__ unsigned Bp[16][132];  // [kpair][n]   (pad 128->132)
  int bz = blockIdx.z;
  int mBase = blockIdx.y * 64;
  int nBase = blockIdx.x * 128;
  const float* Ab = A + (long)bz * sA;
  const float* Bb = Bm + (long)bz * sB;
  float* Cb = C + (long)bz * sC;
  int tid = threadIdx.x;
  int wv = tid >> 5, lane = tid & 31;
  int trow = wv >> 1;
  int cgrp = (wv & 1) * 64;
  int hh = lane >> 4;
  int l15 = lane & 15;

  v8f acc[4];
  acc[0] = {}; acc[1] = {}; acc[2] = {}; acc[3] = {};

  for (int kb = 0; kb < K; kb += 32) {
    // ---- stage A: 64 rows x 16 k-pairs
    for (int i = tid; i < 64 * 16; i += 256) {
      int r = i >> 4, p = i & 15;
      int mg = mBase + r, kg = kb + 2 * p;
      float f0 = 0.f, f1 = 0.f;
      if (mg < M) {
        if (!tA) {
          if (kg + 1 < K) {
            const float2 v2 = *(const float2*)(Ab + (long)mg * lda + kg);
            f0 = v2.x; f1 = v2.y;
          } else if (kg < K) {
            f0 = Ab[(long)mg * lda + kg];
          }
        } else {
          if (kg < K)     f0 = Ab[(long)kg * lda + mg];
          if (kg + 1 < K) f1 = Ab[(long)(kg + 1) * lda + mg];
        }
      }
      PairPK pk; pk.h[0] = (_Float16)f0; pk.h[1] = (_Float16)f1;
      Ap[r][p] = pk.u;
    }
    // ---- stage B: 16 k-pairs x 128 columns
    for (int i = tid; i < 16 * 128; i += 256) {
      int p = i >> 7, c = i & 127;
      int kg = kb + 2 * p, ng = nBase + c;
      float f0 = 0.f, f1 = 0.f;
      if (ng < N) {
        if (tB) {
          if (kg + 1 < K) {
            const float2 v2 = *(const float2*)(Bb + (long)ng * ldb + kg);
            f0 = v2.x; f1 = v2.y;
          } else if (kg < K) {
            f0 = Bb[(long)ng * ldb + kg];
          }
        } else {
          if (kg < K)     f0 = Bb[(long)kg * ldb + ng];
          if (kg + 1 < K) f1 = Bb[(long)(kg + 1) * ldb + ng];
        }
      }
      PairPK pk; pk.h[0] = (_Float16)f0; pk.h[1] = (_Float16)f1;
      Bp[p][c] = pk.u;
    }
    __syncthreads();

    // ---- fragments: pair index p(v,h) = v + (v>=4 ? 4 : 0) + 4*h
    int ml = trow * 16 + l15;
    FragU af;
#pragma unroll
    for (int vv = 0; vv < 8; ++vv) {
      int p = vv + ((vv >= 4) ? 4 : 0) + 4 * hh;
      af.u[vv] = Ap[ml][p];
    }
#pragma unroll
    for (int st = 0; st < 4; ++st) {
      int nl = cgrp + st * 16 + l15;
      FragU bf;
#pragma unroll
      for (int vv = 0; vv < 8; ++vv) {
        int p = vv + ((vv >= 4) ? 4 : 0) + 4 * hh;
        bf.u[vv] = Bp[p][nl];
      }
      acc[st] = __builtin_amdgcn_wmma_f32_16x16x32_f16(false, af.v, false, bf.v,
                                                       (short)0, acc[st], false, false);
    }
    __syncthreads();
  }

#pragma unroll
  for (int st = 0; st < 4; ++st) {
#pragma unroll
    for (int d = 0; d < 8; ++d) {
      int mg = mBase + trow * 16 + d + 8 * hh;
      if (mg >= M) continue;
      float bval = bias ? bias[mg] : 0.f;
      int ng = nBase + cgrp + st * 16 + l15;
      if (ng < N) Cb[(long)mg * ldc + ng] = alpha * acc[st][d] + bval;
    }
  }
}

// ---------------------------------------------------------------- small elementwise kernels
__global__ void scale_ch_kernel(float* __restrict__ x, const float* __restrict__ w, int C, long P) {
  long t = (long)blockIdx.x * 256 + threadIdx.x;
  long tot = (long)Bn * C * P; if (t >= tot) return;
  int c = (int)((t / P) % C);
  x[t] *= w[c];
}
__global__ void maxk_kernel(const float* __restrict__ in, float* __restrict__ out, int C) {
  long t = (long)blockIdx.x * 256 + threadIdx.x;
  long tot = (long)Bn * C * Np; if (t >= tot) return;
  const float* p = in + t * Kn;
  float m = p[0];
  for (int j = 1; j < Kn; ++j) m = fmaxf(m, p[j]);
  out[t] = m;
}
__global__ void softmax_k_kernel(float* __restrict__ x, long rows) {
  long t = (long)blockIdx.x * 256 + threadIdx.x; if (t >= rows) return;
  float* p = x + t * Kn;
  float mx = p[0];
  for (int j = 1; j < Kn; ++j) mx = fmaxf(mx, p[j]);
  float sm = 0.f;
  for (int j = 0; j < Kn; ++j) { float e = expf(p[j] - mx); p[j] = e; sm += e; }
  float inv = 1.f / sm;
  for (int j = 0; j < Kn; ++j) p[j] *= inv;
}
__global__ __launch_bounds__(256)
void softmax_row_kernel(float* __restrict__ att) {
  __shared__ float red[256];
  long row = blockIdx.x;
  float* a = att + row * Np;
  float mx = -3.4e38f;
  for (int i = threadIdx.x; i < Np; i += 256) mx = fmaxf(mx, a[i]);
  red[threadIdx.x] = mx; __syncthreads();
  for (int s = 128; s > 0; s >>= 1) {
    if (threadIdx.x < s) red[threadIdx.x] = fmaxf(red[threadIdx.x], red[threadIdx.x + s]);
    __syncthreads();
  }
  mx = red[0]; __syncthreads();
  float sm = 0.f;
  for (int i = threadIdx.x; i < Np; i += 256) { float e = expf(a[i] - mx); a[i] = e; sm += e; }
  red[threadIdx.x] = sm; __syncthreads();
  for (int s = 128; s > 0; s >>= 1) {
    if (threadIdx.x < s) red[threadIdx.x] += red[threadIdx.x + s];
    __syncthreads();
  }
  float inv = 1.f / red[0];
  for (int i = threadIdx.x; i < Np; i += 256) a[i] *= inv;
}
__global__ void rel_kernel(const float* __restrict__ phi, const float* __restrict__ psi,
                           const float* __restrict__ delta, float* __restrict__ out, long tot) {
  long t = (long)blockIdx.x * 256 + threadIdx.x; if (t >= tot) return;
  long r = t / Kn;
  out[t] = phi[r] - psi[t] + delta[t];
}
__global__ void add_kernel(const float* __restrict__ a, const float* __restrict__ b,
                           float* __restrict__ o, long tot) {
  long t = (long)blockIdx.x * 256 + threadIdx.x; if (t >= tot) return;
  o[t] = a[t] + b[t];
}
__global__ void relu_kernel(float* __restrict__ a, long tot) {
  long t = (long)blockIdx.x * 256 + threadIdx.x; if (t >= tot) return;
  a[t] = fmaxf(a[t], 0.f);
}
__global__ void wsum_kernel(const float* __restrict__ w, const float* __restrict__ f,
                            float* __restrict__ out, long rows) {
  long t = (long)blockIdx.x * 256 + threadIdx.x; if (t >= rows) return;
  const float* wp = w + t * Kn; const float* fp = f + t * Kn;
  float s = 0.f;
  for (int j = 0; j < Kn; ++j) s += wp[j] * fp[j];
  out[t] = s;
}
__global__ void resid_kernel(const float* __restrict__ o2, const float* __restrict__ x,
                             const float* __restrict__ alffa, float* __restrict__ out, long tot) {
  long t = (long)blockIdx.x * 256 + threadIdx.x; if (t >= tot) return;
  out[t] = alffa[0] * o2[t] + x[t];
}
__global__ void dfa_combine_kernel(const float* __restrict__ cb, const float* __restrict__ x0,
                                   const float* __restrict__ x1, float* __restrict__ out, long tot) {
  long t = (long)blockIdx.x * 256 + threadIdx.x; if (t >= tot) return;
  float a = 1.f / (1.f + expf(-cb[t]));
  out[t] = a * x0[t] + (1.f - a) * x1[t];
}
__global__ void copy_ch_kernel(const float* __restrict__ src, float* __restrict__ dst,
                               int C, int Coff, int Ctot) {
  long t = (long)blockIdx.x * 256 + threadIdx.x;
  long tot = (long)Bn * C * Np; if (t >= tot) return;
  int n = (int)(t % Np); long r = t / Np;
  int c = (int)(r % C); int b = (int)(r / C);
  dst[(((long)b * Ctot) + Coff + c) * Np + n] = src[t];
}
__global__ void pool_kernel(const float* __restrict__ f, float* __restrict__ h) {
  int t = blockIdx.x * 256 + threadIdx.x; if (t >= Bn * 1024) return;
  int b = t / 1024, c = t % 1024;
  const float* p = f + ((long)b * 1024 + c) * Np;
  float s = 0.f, m = -3.4e38f;
  for (int i = 0; i < Np; ++i) { s += p[i]; m = fmaxf(m, p[i]); }
  h[b * 2048 + c] = s / (float)Np;
  h[b * 2048 + 1024 + c] = m;
}
__global__ void dense_small_kernel(const float* __restrict__ X, const float* __restrict__ Wm,
                                   const float* __restrict__ bias, float* __restrict__ Y,
                                   int I, int O) {
  int t = blockIdx.x * 256 + threadIdx.x; if (t >= Bn * O) return;
  int b = t / O, o = t % O;
  const float* xp = X + (long)b * I; const float* wp = Wm + (long)o * I;
  float s = bias[o];
  for (int i = 0; i < I; ++i) s += xp[i] * wp[i];
  Y[t] = s;
}
__global__ void bn_rows_kernel(float* __restrict__ X, const float* __restrict__ g,
                               const float* __restrict__ b, int F, int act) {
  int f = blockIdx.x * 256 + threadIdx.x; if (f >= F) return;
  float m = 0.f;
  for (int i = 0; i < Bn; ++i) m += X[i * F + f];
  m /= (float)Bn;
  float v = 0.f;
  for (int i = 0; i < Bn; ++i) { float d = X[i * F + f] - m; v += d * d; }
  v /= (float)Bn;
  float is = rsqrtf(v + EPSBN);
  for (int i = 0; i < Bn; ++i) {
    float y = (X[i * F + f] - m) * is * g[f] + b[f];
    if (act == 1) y = y > 0.f ? y : 0.2f * y;
    X[i * F + f] = y;
  }
}

// ---------------------------------------------------------------- host helpers
static inline unsigned ub(long tot) { return (unsigned)((tot + 255) / 256); }

static void gemm(hipStream_t s, const float* A, const float* B, float* C, const float* bias,
                 int M, int N, int K, long sA, long sB, long sC,
                 int lda, int ldb, int ldc, int tA, int tB, float alpha) {
  dim3 g((unsigned)((N + 127) / 128), (unsigned)((M + 63) / 64), (unsigned)Bn);
  gemm_wmma_kernel<<<g, 256, 0, s>>>(A, B, C, bias, M, N, K, sA, sB, sC, lda, ldb, ldc, tA, tB, alpha);
}
static void run_bn(hipStream_t s, float* x, int C, long P, float* mean, float* istd,
                   const float* g, const float* b, int act) {
  bn_stats_kernel<<<C, 256, 0, s>>>(x, C, P, mean, istd);
  bn_apply_kernel<<<ub((long)Bn * C * P), 256, 0, s>>>(x, C, P, mean, istd, g, b, act);
}
static void deepconv(hipStream_t s, float* gf, float* tmp, float* out, int Ci, int Co,
                     const float* w1, const float* g1, const float* b1,
                     const float* w2, const float* g2, const float* b2,
                     float* mean, float* istd) {
  long P = (long)Np * Kn;
  scale_ch_kernel<<<ub((long)Bn * Ci * P), 256, 0, s>>>(gf, w1, Ci, P);
  run_bn(s, gf, Ci, P, mean, istd, g1, b1, 1);
  gemm(s, w2, gf, tmp, nullptr, Co, (int)P, Ci, 0, (long)Ci * P, (long)Co * P, Ci, (int)P, (int)P, 0, 0, 1.f);
  run_bn(s, tmp, Co, P, mean, istd, g2, b2, 1);
  maxk_kernel<<<ub((long)Bn * Co * Np), 256, 0, s>>>(tmp, out, Co);
}
static void trans2(hipStream_t s, const float* x, float* xs, int C, float* wsA, float* att,
                   const float* wq, const float* bq, const float* wk, const float* bk,
                   const float* wv, const float* bv,
                   const float* f1w, const float* f1b, const float* f2w, const float* f2b,
                   const float* alffa) {
  long szC = (long)Bn * C * Np;
  float* q = wsA; float* kk = wsA + szC; float* v = wsA + 2 * szC;
  float* xr = wsA + 3 * szC; float* o1 = wsA + 4 * szC; float* o2 = wsA + 5 * szC;
  long st = (long)C * Np;
  gemm(s, wq, x, q, bq, C, Np, C, 0, st, st, C, Np, Np, 0, 0, 1.f);
  gemm(s, wk, x, kk, bk, C, Np, C, 0, st, st, C, Np, Np, 0, 0, 1.f);
  gemm(s, wv, x, v, bv, C, Np, C, 0, st, st, C, Np, Np, 0, 0, 1.f);
  float inv = 1.f / sqrtf((float)C);
  gemm(s, q, kk, att, nullptr, Np, Np, C, st, st, (long)Np * Np, Np, Np, Np, 1, 0, inv);
  softmax_row_kernel<<<Bn * Np, 256, 0, s>>>(att);
  gemm(s, v, att, xr, nullptr, C, Np, Np, st, (long)Np * Np, st, Np, Np, Np, 0, 1, 1.f);
  gemm(s, f1w, xr, o1, f1b, C, Np, C, 0, st, st, C, Np, Np, 0, 0, 1.f);
  relu_kernel<<<ub(szC), 256, 0, s>>>(o1, szC);
  gemm(s, f2w, o1, o2, f2b, C, Np, C, 0, st, st, C, Np, Np, 0, 0, 1.f);
  resid_kernel<<<ub(szC), 256, 0, s>>>(o2, x, alffa, xs, szC);
}
static void dfa(hipStream_t s, const float* x0, const float* x1, float* out, int C,
                float* u, float* z, float* c1, float* mean, float* istd,
                const float* fw, const float* fg, const float* fb,
                const float* f1w, const float* f1g, const float* f1b) {
  long szC = (long)Bn * C * Np;
  long st = (long)C * Np;
  add_kernel<<<ub(szC), 256, 0, s>>>(x0, x1, u, szC);
  gemm(s, fw, u, z, nullptr, C, Np, C, 0, st, st, C, Np, Np, 0, 0, 1.f);
  run_bn(s, z, C, Np, mean, istd, fg, fb, 0);
  gemm(s, f1w, z, c1, nullptr, C, Np, C, 0, st, st, C, Np, Np, 0, 0, 1.f);
  run_bn(s, c1, C, Np, mean, istd, f1g, f1b, 0);
  dfa_combine_kernel<<<ub(szC), 256, 0, s>>>(c1, x0, x1, out, szC);
}

// ---------------------------------------------------------------- launch
extern "C" void kernel_launch(void* const* d_in, const int* in_sizes, int n_in,
                              void* d_out, int out_size, void* d_ws, size_t ws_size,
                              hipStream_t stream) {
  (void)in_sizes; (void)n_in; (void)out_size; (void)ws_size;
  const float* x = (const float*)d_in[0];
  const float* cls_b1 = (const float*)d_in[1];
  const float* cls_b2 = (const float*)d_in[2];
  const float* cls_b3 = (const float*)d_in[3];
  const float* cls_bg1 = (const float*)d_in[4];
  const float* cls_bg2 = (const float*)d_in[5];
  const float* cls_g1 = (const float*)d_in[6];
  const float* cls_g2 = (const float*)d_in[7];
  const float* cls_w1 = (const float*)d_in[8];
  const float* cls_w2 = (const float*)d_in[9];
  const float* cls_w3 = (const float*)d_in[10];
  const float *dc_b1[4], *dc_b2[4], *dc_g1[4], *dc_g2[4], *dc_w1[4], *dc_w2[4];
  for (int i = 0; i < 4; ++i) {
    int base = 11 + 6 * i;
    dc_b1[i] = (const float*)d_in[base + 0];
    dc_b2[i] = (const float*)d_in[base + 1];
    dc_g1[i] = (const float*)d_in[base + 2];
    dc_g2[i] = (const float*)d_in[base + 3];
    dc_w1[i] = (const float*)d_in[base + 4];
    dc_w2[i] = (const float*)d_in[base + 5];
  }
  const float *df_f1b[3], *df_f1g[3], *df_f1w[3], *df_fb[3], *df_fg[3], *df_fw[3];
  for (int i = 0; i < 3; ++i) {
    int base = 35 + 6 * i;
    df_f1b[i] = (const float*)d_in[base + 0];
    df_f1g[i] = (const float*)d_in[base + 1];
    df_f1w[i] = (const float*)d_in[base + 2];
    df_fb[i]  = (const float*)d_in[base + 3];
    df_fg[i]  = (const float*)d_in[base + 4];
    df_fw[i]  = (const float*)d_in[base + 5];
  }
  const float* out_b  = (const float*)d_in[53];
  const float* out_bb = (const float*)d_in[54];
  const float* out_g  = (const float*)d_in[55];
  const float* out_w  = (const float*)d_in[56];
  const float* pt_alb = (const float*)d_in[57];
  const float* pt_alw = (const float*)d_in[58];
  const float* pt_bgb = (const float*)d_in[59];
  const float* pt_bgg = (const float*)d_in[60];
  const float* pt_btb = (const float*)d_in[61];
  const float* pt_btg = (const float*)d_in[62];
  const float* pt_g1b = (const float*)d_in[63];
  const float* pt_g1w = (const float*)d_in[64];
  const float* pt_g2b = (const float*)d_in[65];
  const float* pt_g2w = (const float*)d_in[66];
  const float* pt_phb = (const float*)d_in[67];
  const float* pt_phw = (const float*)d_in[68];
  const float* pt_psb = (const float*)d_in[69];
  const float* pt_psw = (const float*)d_in[70];
  const float* pt_t1b = (const float*)d_in[71];
  const float* pt_t1w = (const float*)d_in[72];
  const float* pt_t2b = (const float*)d_in[73];
  const float* pt_t2w = (const float*)d_in[74];
  const float *tr_alffa[3], *tr_bk[3], *tr_bq[3], *tr_bv[3], *tr_f1b[3], *tr_f1w[3],
              *tr_f2b[3], *tr_f2w[3], *tr_wk[3], *tr_wq[3], *tr_wv[3];
  for (int i = 0; i < 3; ++i) {
    int base = 75 + 11 * i;
    tr_alffa[i] = (const float*)d_in[base + 0];
    tr_bk[i] = (const float*)d_in[base + 1];
    tr_bq[i] = (const float*)d_in[base + 2];
    tr_bv[i] = (const float*)d_in[base + 3];
    tr_f1b[i] = (const float*)d_in[base + 4];
    tr_f1w[i] = (const float*)d_in[base + 5];
    tr_f2b[i] = (const float*)d_in[base + 6];
    tr_f2w[i] = (const float*)d_in[base + 7];
    tr_wk[i] = (const float*)d_in[base + 8];
    tr_wq[i] = (const float*)d_in[base + 9];
    tr_wv[i] = (const float*)d_in[base + 10];
  }

  float* W = (float*)d_ws;
  const long SZK256 = 83886080L;          // [8,256,2048,20]
  float* A0 = W;
  float* A1 = W + SZK256;
  float* A2 = W + 2 * SZK256;             // 41,943,040 floats
  long prs = 2 * SZK256 + 41943040L;
  float* xd   = W + prs; prs += (long)Bn * 8 * Np;
  float* x1   = W + prs; prs += (long)Bn * 64 * Np;
  float* x1t  = W + prs; prs += (long)Bn * 64 * Np;
  float* x2   = W + prs; prs += (long)Bn * 64 * Np;
  float* x2t  = W + prs; prs += (long)Bn * 64 * Np;
  float* x3   = W + prs; prs += (long)Bn * 128 * Np;
  float* x3t  = W + prs; prs += (long)Bn * 128 * Np;
  float* x4   = W + prs; prs += (long)Bn * 256 * Np;
  float* x4t  = W + prs; prs += (long)Bn * 256 * Np;
  float* fcat = W + prs; prs += (long)Bn * 512 * Np;
  float* hpool = W + prs; prs += (long)Bn * 2048;
  float* hid1  = W + prs; prs += (long)Bn * 512;
  float* hid2  = W + prs; prs += (long)Bn * 256;
  float* bmean = W + prs; prs += 1024;
  float* bistd = W + prs; prs += 1024;
  int* idxk = (int*)(W + prs); prs += (long)Bn * Np * Kn;
  int* idx3 = (int*)(W + prs); prs += (long)Bn * Np * 3;

  long Pk = (long)Np * Kn;                // 40960

  // ---- geometric point descriptor
  knn_kernel<<<Bn * Np, 256, 0, stream>>>(x, 3, 3, idx3);
  geom_desc_kernel<<<ub((long)Bn * Np), 256, 0, stream>>>(x, idx3, xd);

  // ---- stage 1: deepconv dc1 (xd -> x1, 16 -> 64)
  knn_kernel<<<Bn * Np, 256, 0, stream>>>(xd, 8, Kn, idxk);
  graph_feature_kernel<<<ub((long)Bn * 8 * Np * Kn), 256, 0, stream>>>(xd, idxk, A0, 8);
  deepconv(stream, A0, A1, x1, 16, 64, dc_w1[0], dc_g1[0], dc_b1[0], dc_w2[0], dc_g2[0], dc_b2[0], bmean, bistd);

  // ---- point transformer (xyz, x1) -> x1t
  knn_kernel<<<Bn * Np, 256, 0, stream>>>(x, 3, Kn, idxk);
  gather_kernel<<<ub((long)Bn * 3 * Np * Kn), 256, 0, stream>>>(x, idxk, A0, 3, 1);   // pos
  gemm(stream, pt_t1w, A0, A1, pt_t1b, 64, (int)Pk, 3, 0, 3 * Pk, 64 * Pk, 3, (int)Pk, (int)Pk, 0, 0, 1.f);
  gemm(stream, pt_t2w, A1, A0, pt_t2b, 64, (int)Pk, 64, 0, 64 * Pk, 64 * Pk, 64, (int)Pk, (int)Pk, 0, 0, 1.f);
  run_bn(stream, A0, 64, Pk, bmean, bistd, pt_btg, pt_btb, 2);                        // delta @ A0
  long totxj = (long)Bn * 64 * Np * Kn;
  gather_kernel<<<ub(totxj), 256, 0, stream>>>(x1, idxk, A2, 64, 0);                  // xj @ A2
  gemm(stream, pt_phw, x1, A1, pt_phb, 64, Np, 64, 0, (long)64 * Np, (long)64 * Np, 64, Np, Np, 0, 0, 1.f); // phi
  float* psi = A1 + 2097152;
  gemm(stream, pt_psw, A2, psi, pt_psb, 64, (int)Pk, 64, 0, 64 * Pk, 64 * Pk, 64, (int)Pk, (int)Pk, 0, 0, 1.f);
  rel_kernel<<<ub(totxj), 256, 0, stream>>>(A1, psi, A0, psi, totxj);                 // rel in place
  float* g1buf = A2 + 20971520;
  gemm(stream, pt_g1w, psi, g1buf, pt_g1b, 64, (int)Pk, 64, 0, 64 * Pk, 64 * Pk, 64, (int)Pk, (int)Pk, 0, 0, 1.f);
  gemm(stream, pt_g2w, g1buf, psi, pt_g2b, 64, (int)Pk, 64, 0, 64 * Pk, 64 * Pk, 64, (int)Pk, (int)Pk, 0, 0, 1.f);
  run_bn(stream, psi, 64, Pk, bmean, bistd, pt_bgg, pt_bgb, 2);
  softmax_k_kernel<<<ub((long)Bn * 64 * Np), 256, 0, stream>>>(psi, (long)Bn * 64 * Np);
  gemm(stream, pt_alw, A2, g1buf, pt_alb, 64, (int)Pk, 64, 0, 64 * Pk, 64 * Pk, 64, (int)Pk, (int)Pk, 0, 0, 1.f);
  add_kernel<<<ub(totxj), 256, 0, stream>>>(g1buf, A0, g1buf, totxj);                 // f = alw(xj)+delta
  wsum_kernel<<<ub((long)Bn * 64 * Np), 256, 0, stream>>>(psi, g1buf, x1t, (long)Bn * 64 * Np);

  // ---- stage 2: dc2 + trans2 + dfa2
  knn_kernel<<<Bn * Np, 256, 0, stream>>>(x1t, 64, Kn, idxk);
  graph_feature_kernel<<<ub((long)Bn * 64 * Np * Kn), 256, 0, stream>>>(x1t, idxk, A0, 64);
  deepconv(stream, A0, A1, x2, 128, 64, dc_w1[1], dc_g1[1], dc_b1[1], dc_w2[1], dc_g2[1], dc_b2[1], bmean, bistd);
  {
    long szC = (long)Bn * 64 * Np;
    float* xs = A0 + 6 * szC;
    trans2(stream, x2, xs, 64, A0, A1, tr_wq[0], tr_bq[0], tr_wk[0], tr_bk[0], tr_wv[0], tr_bv[0],
           tr_f1w[0], tr_f1b[0], tr_f2w[0], tr_f2b[0], tr_alffa[0]);
    dfa(stream, x2, xs, x2t, 64, A1, A1 + szC, A1 + 2 * szC, bmean, bistd,
        df_fw[0], df_fg[0], df_fb[0], df_f1w[0], df_f1g[0], df_f1b[0]);
  }

  // ---- stage 3: dc3 + trans2 + dfa3
  knn_kernel<<<Bn * Np, 256, 0, stream>>>(x2t, 64, Kn, idxk);
  graph_feature_kernel<<<ub((long)Bn * 64 * Np * Kn), 256, 0, stream>>>(x2t, idxk, A0, 64);
  deepconv(stream, A0, A1, x3, 128, 128, dc_w1[2], dc_g1[2], dc_b1[2], dc_w2[2], dc_g2[2], dc_b2[2], bmean, bistd);
  {
    long szC = (long)Bn * 128 * Np;
    float* xs = A0 + 6 * szC;
    trans2(stream, x3, xs, 128, A0, A1, tr_wq[1], tr_bq[1], tr_wk[1], tr_bk[1], tr_wv[1], tr_bv[1],
           tr_f1w[1], tr_f1b[1], tr_f2w[1], tr_f2b[1], tr_alffa[1]);
    dfa(stream, x3, xs, x3t, 128, A1, A1 + szC, A1 + 2 * szC, bmean, bistd,
        df_fw[1], df_fg[1], df_fb[1], df_f1w[1], df_f1g[1], df_f1b[1]);
  }

  // ---- stage 4: dc4 + trans2 + dfa4
  knn_kernel<<<Bn * Np, 256, 0, stream>>>(x3t, 128, Kn, idxk);
  graph_feature_kernel<<<ub((long)Bn * 128 * Np * Kn), 256, 0, stream>>>(x3t, idxk, A0, 128);
  deepconv(stream, A0, A1, x4, 256, 256, dc_w1[3], dc_g1[3], dc_b1[3], dc_w2[3], dc_g2[3], dc_b2[3], bmean, bistd);
  {
    long szC = (long)Bn * 256 * Np;
    float* xs = A0 + 6 * szC;
    trans2(stream, x4, xs, 256, A0, A1, tr_wq[2], tr_bq[2], tr_wk[2], tr_bk[2], tr_wv[2], tr_bv[2],
           tr_f1w[2], tr_f1b[2], tr_f2w[2], tr_f2b[2], tr_alffa[2]);
    dfa(stream, x4, xs, x4t, 256, A1, A1 + szC, A1 + 2 * szC, bmean, bistd,
        df_fw[2], df_fg[2], df_fb[2], df_f1w[2], df_f1g[2], df_f1b[2]);
  }

  // ---- head
  copy_ch_kernel<<<ub((long)Bn * 64 * Np), 256, 0, stream>>>(x1t, fcat, 64, 0, 512);
  copy_ch_kernel<<<ub((long)Bn * 64 * Np), 256, 0, stream>>>(x2t, fcat, 64, 64, 512);
  copy_ch_kernel<<<ub((long)Bn * 128 * Np), 256, 0, stream>>>(x3t, fcat, 128, 128, 512);
  copy_ch_kernel<<<ub((long)Bn * 256 * Np), 256, 0, stream>>>(x4t, fcat, 256, 256, 512);
  gemm(stream, out_w, fcat, A0, out_b, 1024, Np, 512, 0, (long)512 * Np, (long)1024 * Np, 512, Np, Np, 0, 0, 1.f);
  run_bn(stream, A0, 1024, Np, bmean, bistd, out_g, out_bb, 1);
  pool_kernel<<<ub((long)Bn * 1024), 256, 0, stream>>>(A0, hpool);
  dense_small_kernel<<<ub((long)Bn * 512), 256, 0, stream>>>(hpool, cls_w1, cls_b1, hid1, 2048, 512);
  bn_rows_kernel<<<ub(512), 256, 0, stream>>>(hid1, cls_g1, cls_bg1, 512, 1);
  dense_small_kernel<<<ub((long)Bn * 256), 256, 0, stream>>>(hid1, cls_w2, cls_b2, hid2, 512, 256);
  bn_rows_kernel<<<ub(256), 256, 0, stream>>>(hid2, cls_g2, cls_bg2, 256, 1);
  dense_small_kernel<<<ub((long)Bn * 40), 256, 0, stream>>>(hid2, cls_w3, cls_b3, (float*)d_out, 256, 40);
}
